// FrameVQVAE_17669495455858
// MI455X (gfx1250) — matrix operand order repaired and checked by reading, not compiled, not run on hardware
//
#include <hip/hip_runtime.h>
#include <hip/hip_bf16.h>
#include <math.h>

typedef _Float16 f16;
typedef __attribute__((ext_vector_type(16))) _Float16 v16h;
typedef __attribute__((ext_vector_type(8)))  float    v8f;
typedef __attribute__((ext_vector_type(4)))  unsigned int v4u;
typedef __attribute__((ext_vector_type(8)))  int      v8i;
typedef __attribute__((ext_vector_type(4)))  int      v4i;

#define BATCH      131072
#define XHAT_OFF   0
#define ZE_OFF     8388608    // B*64
#define ZQ_OFF     12582912   // +B*32
#define IDX_OFF    16777216   // +B*32
#define QL_OFF     16908288   // +B
#define PP_OFF     16908289

// packed f16 weight offsets in halfs (A-fragment layout of W^T); frag = 512 halfs = 1KB
#define OFF_WE1 0
#define OFF_WE2 131072
#define OFF_WE3 196608
#define OFF_WD1 204800
#define OFF_WD2 278528
#define OFF_WD3 344064
#define OFF_CB  360448
#define WS_F32_BYTE 753664    // end of f16 region (376832 halfs)

__device__ __forceinline__ v8f zero8() {
  v8f z = {0.f,0.f,0.f,0.f,0.f,0.f,0.f,0.f};
  return z;
}

__device__ __forceinline__ v8f wmma_f16(v16h a, v16h b, v8f c) {
  return __builtin_amdgcn_wmma_f32_16x16x32_f16(false, a, false, b, (short)0, c, false, false);
}

__device__ __forceinline__ unsigned int ldsoff(const void* p) {
  // generic pointer to __shared__: low 32 bits are the workgroup-relative LDS byte offset
  return (unsigned int)(uintptr_t)p;
}

// ---- Tensor Data Mover: DMA `bytes` (multiple of 8) from global into LDS ----
// D# per CDNA5 ISA: 1-D tile of 8-byte elements, no
// multicast, no padding/iteration.  Tracked by TENSORcnt.
__device__ __forceinline__ void tdm_load_chunk(const f16* gsrc, unsigned int lds_byte,
                                               unsigned int bytes) {
  unsigned long long ga = (unsigned long long)(uintptr_t)gsrc;
  unsigned int elems = bytes >> 3;                       // 8B elements (<= 2048)
  v4u g0;
  g0[0] = 1u;                                            // count=1, user descriptor
  g0[1] = lds_byte;                                      // lds_addr
  g0[2] = (unsigned int)ga;                              // global_addr[31:0]
  g0[3] = ((unsigned int)(ga >> 32) & 0x01FFFFFFu) | (2u << 30);  // addr[56:32] | type=2
  v8i g1;
  g1[0] = (int)(3u << 16);                               // wg_mask=0, data_size=3 (8B)
  g1[1] = (int)((elems & 0xFFFFu) << 16);                // tensor_dim0[15:0]
  g1[2] = (int)(((elems >> 16) & 0xFFFFu) | (1u << 16)); // dim0[31:16] | tensor_dim1=1
  g1[3] = (int)((elems & 0xFFFFu) << 16);                // tile_dim0 = elems
  g1[4] = 1;                                             // tile_dim1 = 1
  g1[5] = (int)elems;                                    // tensor_dim0_stride lo
  g1[6] = 0;
  g1[7] = 0;
  v4i g2 = {0,0,0,0};
  v4i g3 = {0,0,0,0};
  v8i g4 = {0,0,0,0,0,0,0,0};
  __builtin_amdgcn_tensor_load_to_lds(g0, g1, g2, g3, g4, 0);
}

// B-fragment (K x 16 batch) straight from a row-major f32 activation matrix.
__device__ __forceinline__ v16h load_actfrag(const float* base, int ld, int k0, int lane) {
  const float* p = base + (size_t)(lane & 15) * ld + k0 + ((lane & 16) ? 16 : 0);
  const float4* p4 = (const float4*)p;
  float4 x0 = p4[0], x1 = p4[1], x2 = p4[2], x3 = p4[3];
  float v[16] = {x0.x,x0.y,x0.z,x0.w, x1.x,x1.y,x1.z,x1.w,
                 x2.x,x2.y,x2.z,x2.w, x3.x,x3.y,x3.z,x3.w};
  v16h f;
#pragma unroll
  for (int i = 0; i < 16; ++i) f[i] = (f16)v[i];
  return f;
}

__device__ __forceinline__ void bias_act(v8f d, const float* bias, int fbase, bool hi,
                                         bool relu, float (&h)[8]) {
  const float4* bp = (const float4*)(bias + fbase + (hi ? 8 : 0));
  float4 b0 = bp[0], b1 = bp[1];
  float bb[8] = {b0.x,b0.y,b0.z,b0.w, b1.x,b1.y,b1.z,b1.w};
#pragma unroll
  for (int r = 0; r < 8; ++r) {
    float v = d[r] + bb[r];
    h[r] = relu ? fmaxf(v, 0.0f) : v;
  }
}

// D-tile pair -> next layer's B-fragment; only cross-lane motion is shfl_xor(.,16).
__device__ __forceinline__ v16h make_bfrag(const float (&h0)[8], const float (&h1)[8], bool hi) {
  float o0[8], o1[8];
#pragma unroll
  for (int r = 0; r < 8; ++r) {
    o0[r] = __shfl_xor(h0[r], 16, 32);
    o1[r] = __shfl_xor(h1[r], 16, 32);
  }
  v16h f;
#pragma unroll
  for (int h = 0; h < 16; ++h) {
    float lo  = (h < 8) ? h0[h] : o0[h - 8];
    float hiv = (h < 8) ? o1[h] : h1[h - 8];
    f[h] = (f16)(hi ? hiv : lo);
  }
  return f;
}

// One MLP layer: out^T = W^T * in^T.  Weights stream global->LDS via TDM
// (double buffered, one 16-feature tile per chunk), all 8 waves consume the
// same chunk in lockstep.  emit(mt, h) receives the f32 D-tile per mt.
template<int KT, int MT, class F>
__device__ __forceinline__ void mlp_layer_tdm(const f16* wsrc, const float* bias,
                                              const v16h (&Bin)[KT], f16 (&wbuf)[2][8192],
                                              int lane, int wave, bool hi, bool relu,
                                              F&& emit) {
  __syncthreads();                       // previous consumers of wbuf are done
  if (wave == 0) tdm_load_chunk(wsrc, ldsoff(&wbuf[0][0]), KT * 1024u);
#pragma unroll
  for (int mt = 0; mt < MT; ++mt) {
    if (wave == 0) __builtin_amdgcn_s_wait_tensorcnt(0);   // chunk mt landed
    __syncthreads();                                       // visible to all waves
    if (wave == 0 && (mt + 1) < MT)
      tdm_load_chunk(wsrc + (size_t)(mt + 1) * KT * 512, ldsoff(&wbuf[(mt + 1) & 1][0]),
                     KT * 1024u);
    const v16h* wb = (const v16h*)&wbuf[mt & 1][0];
    v8f d = zero8();
#pragma unroll
    for (int kt = 0; kt < KT; ++kt)
      d = wmma_f16(wb[kt * 32 + lane], Bin[kt], d);
    float h[8];
    bias_act(d, bias, mt * 16, hi, relu, h);
    emit(mt, h);
  }
}

// ---------------- prep: pack W^T (or codebook) into A-fragment layout ----------------
__global__ void pack_w_kernel(const float* __restrict__ src, f16* __restrict__ dst,
                              int KT, int MT, int stride, int transpose, int total) {
  int t = blockIdx.x * blockDim.x + threadIdx.x;
  if (t >= total) return;
  int h  = t & 15;
  int l  = (t >> 4) & 31;
  int fr = t >> 9;
  int kt = fr % KT;
  int mt = fr / KT;
  int M = mt * 16 + (l & 15);
  int K = kt * 32 + ((l & 16) ? 16 : 0) + h;
  float v = transpose ? src[(size_t)K * stride + M] : src[(size_t)M * stride + K];
  dst[t] = (f16)v;
}

__global__ void cbnorm_kernel(const float* __restrict__ cb, float* __restrict__ cbn) {
  int n = blockIdx.x * blockDim.x + threadIdx.x;
  if (n < 512) {
    float s = 0.f;
#pragma unroll
    for (int j = 0; j < 32; ++j) { float v = cb[n * 32 + j]; s += v * v; }
    cbn[n] = s;
  }
}

// ---------------- fused VQ-VAE forward: 16 rows/wave, 8 waves/block ----------
__global__ __launch_bounds__(256) void vqvae_fused(
    const float* __restrict__ enc, const float* __restrict__ cond,
    const f16* __restrict__ wp,
    const float* __restrict__ be1, const float* __restrict__ be2, const float* __restrict__ be3,
    const float* __restrict__ bd1, const float* __restrict__ bd2, const float* __restrict__ bd3,
    const float* __restrict__ cb, const float* __restrict__ cbn,
    float* __restrict__ hist, float* __restrict__ errsum, float* __restrict__ out) {
  __shared__ f16 WBUF[2][8192];          // 2 x 16KB weight-chunk double buffer

  const int lane = threadIdx.x & 31;
  const int wave = threadIdx.x >> 5;
  const int rowbase = blockIdx.x * 128 + wave * 16;
  const bool hi = (lane & 16) != 0;
  const int bcol = lane & 15;

  // ---- Encoder L1: relu(X W1 + b1) as W1^T X^T ----
  v16h B1[16];
#pragma unroll
  for (int kt = 0; kt < 16; ++kt)
    B1[kt] = load_actfrag(enc + (size_t)rowbase * 512, 512, kt * 32, lane);

  v16h B2[8];
  {
    float hprev[8];
    mlp_layer_tdm<16, 16>(wp + OFF_WE1, be1, B1, WBUF, lane, wave, hi, true,
      [&](int mt, float (&h)[8]) {
        if (mt & 1) { B2[mt >> 1] = make_bfrag(hprev, h, hi); }
        else {
#pragma unroll
          for (int r = 0; r < 8; ++r) hprev[r] = h[r];
        }
      });
  }
  v16h B3[8];
  {
    float hprev[8];
    mlp_layer_tdm<8, 16>(wp + OFF_WE2, be2, B2, WBUF, lane, wave, hi, true,
      [&](int mt, float (&h)[8]) {
        if (mt & 1) { B3[mt >> 1] = make_bfrag(hprev, h, hi); }
        else {
#pragma unroll
          for (int r = 0; r < 8; ++r) hprev[r] = h[r];
        }
      });
  }

  // ---- Encoder L3 -> z_e (16x32), keep f32 ----
  float z0[8], z1[8];
  mlp_layer_tdm<8, 2>(wp + OFF_WE3, be3, B3, WBUF, lane, wave, hi, false,
    [&](int mt, float (&h)[8]) {
#pragma unroll
      for (int r = 0; r < 8; ++r) { if (mt == 0) z0[r] = h[r]; else z1[r] = h[r]; }
    });
  {
    float* zo = out + ZE_OFF + (size_t)(rowbase + bcol) * 32;
#pragma unroll
    for (int r = 0; r < 8; ++r) {
      zo[(hi ? 8 : 0) + r]      = z0[r];
      zo[16 + (hi ? 8 : 0) + r] = z1[r];
    }
  }
  v16h Bz = make_bfrag(z0, z1, hi);

  // ---- VQ: argmin_n (|c_n|^2 - 2 z.c_n); codebook tiles via TDM too ----
  float minv = 3.4e38f;
  int   mini = 0;
  __syncthreads();
  if (wave == 0) tdm_load_chunk(wp + OFF_CB, ldsoff(&WBUF[0][0]), 16 * 1024u);
#pragma unroll
  for (int half = 0; half < 2; ++half) {
    if (wave == 0) __builtin_amdgcn_s_wait_tensorcnt(0);
    __syncthreads();
    if (wave == 0 && half == 0)
      tdm_load_chunk(wp + OFF_CB + 16 * 512, ldsoff(&WBUF[1][0]), 16 * 1024u);
    const v16h* wb = (const v16h*)&WBUF[half][0];
#pragma unroll
    for (int ct2 = 0; ct2 < 16; ++ct2) {
      int ct = half * 16 + ct2;
      v8f d = wmma_f16(wb[ct2 * 32 + lane], Bz, zero8());
      const float4* np = (const float4*)(cbn + ct * 16 + (hi ? 8 : 0));
      float4 n0 = np[0], n1 = np[1];
      float nn[8] = {n0.x,n0.y,n0.z,n0.w, n1.x,n1.y,n1.z,n1.w};
#pragma unroll
      for (int r = 0; r < 8; ++r) {
        float s = nn[r] - 2.0f * d[r];
        int code = ct * 16 + (hi ? 8 : 0) + r;
        if (s < minv) { minv = s; mini = code; }
      }
    }
  }
  {
    float ov = __shfl_xor(minv, 16, 32);
    int   oi = __shfl_xor(mini, 16, 32);
    if (ov < minv || (ov == minv && oi < mini)) { minv = ov; mini = oi; }
  }

  // ---- gather z_q row ----
  float qa[32];
  {
    const float4* qp = (const float4*)(cb + (size_t)mini * 32);
#pragma unroll
    for (int j = 0; j < 8; ++j) {
      float4 q = qp[j];
      qa[4 * j] = q.x; qa[4 * j + 1] = q.y; qa[4 * j + 2] = q.z; qa[4 * j + 3] = q.w;
    }
  }
  if (lane < 16) {
    float* qo = out + ZQ_OFF + (size_t)(rowbase + bcol) * 32;
#pragma unroll
    for (int j = 0; j < 32; ++j) qo[j] = qa[j];
    out[IDX_OFF + rowbase + bcol] = (float)mini;
    atomicAdd(&hist[mini], 1.0f);
  }
  // ---- loss accumulation: sum (z_q - z_e)^2 ----
  {
    float e = 0.0f;
#pragma unroll
    for (int r = 0; r < 8; ++r) {
      float q0 = hi ? qa[8 + r]  : qa[r];
      float q1 = hi ? qa[24 + r] : qa[16 + r];
      float e0 = q0 - z0[r], e1 = q1 - z1[r];
      e += e0 * e0 + e1 * e1;
    }
#pragma unroll
    for (int off = 16; off > 0; off >>= 1) e += __shfl_xor(e, off, 32);
    if (lane == 0) atomicAdd(errsum, e);
  }

  // ---- Decoder on [z_q, cond] (K = 288) ----
  v16h Bd[9];
  {
    v16h f;
#pragma unroll
    for (int h = 0; h < 16; ++h) f[h] = (f16)(hi ? qa[16 + h] : qa[h]);
    Bd[0] = f;
  }
#pragma unroll
  for (int kt = 1; kt < 9; ++kt)
    Bd[kt] = load_actfrag(cond + (size_t)rowbase * 256, 256, (kt - 1) * 32, lane);

  v16h D1[8];
  {
    float hprev[8];
    mlp_layer_tdm<9, 16>(wp + OFF_WD1, bd1, Bd, WBUF, lane, wave, hi, true,
      [&](int mt, float (&h)[8]) {
        if (mt & 1) { D1[mt >> 1] = make_bfrag(hprev, h, hi); }
        else {
#pragma unroll
          for (int r = 0; r < 8; ++r) hprev[r] = h[r];
        }
      });
  }
  v16h D2[8];
  {
    float hprev[8];
    mlp_layer_tdm<8, 16>(wp + OFF_WD2, bd2, D1, WBUF, lane, wave, hi, true,
      [&](int mt, float (&h)[8]) {
        if (mt & 1) { D2[mt >> 1] = make_bfrag(hprev, h, hi); }
        else {
#pragma unroll
          for (int r = 0; r < 8; ++r) hprev[r] = h[r];
        }
      });
  }
  // final layer: 256 -> 64, bias, no relu, store x_hat
  mlp_layer_tdm<8, 4>(wp + OFF_WD3, bd3, D2, WBUF, lane, wave, hi, false,
    [&](int mt, float (&h)[8]) {
      float* xo = out + XHAT_OFF + (size_t)(rowbase + bcol) * 64 + mt * 16 + (hi ? 8 : 0);
#pragma unroll
      for (int r = 0; r < 8; ++r) xo[r] = h[r];
    });
}

// ---------------- finalize: perplexity + quant_loss scalars ----------------
__global__ void finalize_kernel(const float* __restrict__ hist, const float* __restrict__ errsum,
                                float* __restrict__ out) {
  __shared__ float red[512];
  int t = threadIdx.x;
  float p = hist[t] * (1.0f / (float)BATCH);
  red[t] = p * logf(p + 1e-10f);
  __syncthreads();
  for (int s = 256; s > 0; s >>= 1) {
    if (t < s) red[t] += red[t + s];
    __syncthreads();
  }
  if (t == 0) {
    out[QL_OFF] = 1.25f * errsum[0] / ((float)BATCH * 32.0f);  // (1+beta)*mse
    out[PP_OFF] = expf(-red[0]);
  }
}

extern "C" void kernel_launch(void* const* d_in, const int* in_sizes, int n_in,
                              void* d_out, int out_size, void* d_ws, size_t ws_size,
                              hipStream_t stream) {
  (void)in_sizes; (void)n_in; (void)out_size; (void)ws_size;
  const float* enc = (const float*)d_in[0];
  const float* cnd = (const float*)d_in[1];
  const float* We1 = (const float*)d_in[2];  const float* be1 = (const float*)d_in[3];
  const float* We2 = (const float*)d_in[4];  const float* be2 = (const float*)d_in[5];
  const float* We3 = (const float*)d_in[6];  const float* be3 = (const float*)d_in[7];
  const float* Wd1 = (const float*)d_in[8];  const float* bd1 = (const float*)d_in[9];
  const float* Wd2 = (const float*)d_in[10]; const float* bd2 = (const float*)d_in[11];
  const float* Wd3 = (const float*)d_in[12]; const float* bd3 = (const float*)d_in[13];
  const float* cb  = (const float*)d_in[14];
  float* out = (float*)d_out;

  f16*   wp   = (f16*)d_ws;
  float* f32w = (float*)((char*)d_ws + WS_F32_BYTE);
  float* cbn  = f32w;          // 512
  float* hist = f32w + 512;    // 512
  float* errs = f32w + 1024;   // 1

  (void)hipMemsetAsync(hist, 0, 513 * sizeof(float), stream);

  struct Pk { const float* src; int off, KT, MT, stride, tr; };
  const Pk pk[7] = {
    {We1, OFF_WE1, 16, 16, 256, 1},
    {We2, OFF_WE2,  8, 16, 256, 1},
    {We3, OFF_WE3,  8,  2,  32, 1},
    {Wd1, OFF_WD1,  9, 16, 256, 1},
    {Wd2, OFF_WD2,  8, 16, 256, 1},
    {Wd3, OFF_WD3,  8,  4,  64, 1},
    {cb,  OFF_CB,   1, 32,  32, 0},
  };
  for (int i = 0; i < 7; ++i) {
    int total = pk[i].KT * pk[i].MT * 512;
    pack_w_kernel<<<(total + 255) / 256, 256, 0, stream>>>(
        pk[i].src, wp + pk[i].off, pk[i].KT, pk[i].MT, pk[i].stride, pk[i].tr, total);
  }
  cbnorm_kernel<<<2, 256, 0, stream>>>(cb, cbn);

  vqvae_fused<<<BATCH / 128, 256, 0, stream>>>(
      enc, cnd, wp, be1, be2, be3, bd1, bd2, bd3, cb, cbn, hist, errs, out);

  finalize_kernel<<<1, 512, 0, stream>>>(hist, errs, out);
}
